// Transformer_58566174048399
// MI455X (gfx1250) — compile-verified, matrix operand order checked by reading, not run on hardware
//
#include <hip/hip_runtime.h>
#include <math.h>

// ---------------------------------------------------------------------------
// Sizes from the reference
// ---------------------------------------------------------------------------
#define N_NODES   16000
#define N_EDGES   80000
#define N_GRAPHS  64
#define N_BASIS   10
#define NODE_DIM  64           // Z_DIM(48) + MOL_DIM(16)
#define MUL       16
#define MID       32
#define N_TILES   (N_EDGES / 16)   // 5000, exact

// constants (scales folded where noted)
#define SILU_NORM     1.679177f
#define SMOOTH_C      8.4335731f     // 1.14136 * e^2
#define SQRT10        3.16227766f
#define INV_SQRT10    0.316227766f
#define S3            1.73205081f
#define QK_NORM       0.0441941738f  // 1/sqrt(2*16*16)
#define B_SCALE       0.0220970869f  // 1/sqrt(MID*NODE_DIM) folded into packed W2

// ---------------------------------------------------------------------------
// WMMA vector types (CDNA5 / gfx1250, wave32)
// ---------------------------------------------------------------------------
typedef __attribute__((ext_vector_type(16))) __bf16 bf16x16;  // 16x32 bf16 A/B fragment (8 VGPRs)
typedef __attribute__((ext_vector_type(8)))  float  f32x8;    // 16x16 f32 C/D fragment (8 VGPRs)

// ---------------------------------------------------------------------------
// helpers
// ---------------------------------------------------------------------------
__device__ __forceinline__ float sus(float x) {
    return x > 0.f ? __expf(-1.f / x) : 0.f;
}

__device__ __forceinline__ unsigned f2bf(float f) {  // round-to-nearest-even bf16
    union { float f; unsigned u; } v; v.f = f;
    unsigned r = v.u + 0x7fffu + ((v.u >> 16) & 1u);
    return r >> 16;
}

// K-pair held by (vgpr v, lane-half hi) of a 16-bit 16x32 WMMA operand
// (ISA 7.12.2: VGPR0..3 -> K {0,1}..{6,7} (+8 for hi lanes); VGPR4..7 -> K 16..23 (+8)).
__device__ __host__ __forceinline__ int frag_k0(int v, int hi) {
    int base = (v < 4) ? (2 * v) : (2 * v + 8);
    return base + 8 * hi;      // pair is {K0, K0+1}
}

// ---------------------------------------------------------------------------
// workspace layout (float units)
// ---------------------------------------------------------------------------
#define OFF_FEAT  ((size_t)0)                 // 16000*64
#define OFF_QA    ((size_t)1024000)           // 16000*16
#define OFF_CUT   ((size_t)1280000)           // 80000
#define OFF_SH1   ((size_t)1360000)           // 80000*3
#define OFF_HAK   ((size_t)1600000)           // u32: 5000*32*8  (packed bf16 A frags, k)
#define OFF_HAV   ((size_t)2880000)           // u32: 5000*32*8  (v)
#define OFF_BPK   ((size_t)4160000)           // u32: 64 *32*8   (packed B frags for k0)
#define OFF_BPV   ((size_t)4176384)           // u32: 128*32*8   (packed B frags for v0,v1)
#define OFF_EXPV  ((size_t)4209152)           // 80000
#define OFF_V0    ((size_t)4289152)           // 80000*16
#define OFF_V1    ((size_t)5569152)           // 80000*16
#define OFF_Z     ((size_t)6849152)           // 16000          (zeroed)
#define OFF_NODE  ((size_t)6865152)           // 16000*64       (zeroed)
#define OFF_GSUM  ((size_t)7889152)           // 64*64          (zeroed)
#define OFF_GCNT  ((size_t)7893248)           // 64             (zeroed)
#define WS_TOTAL  ((size_t)7893312)           // ~30.1 MB

// ---------------------------------------------------------------------------
// Kernel 1: node features + qa = (feat @ Wq / sqrt(64)) @ Wa
// ---------------------------------------------------------------------------
__global__ void node_kernel(const float* __restrict__ z_table,
                            const float* __restrict__ mol_table,
                            const float* __restrict__ Wq,
                            const float* __restrict__ Wa,
                            const int*   __restrict__ x,
                            const int*   __restrict__ mol_id,
                            float* __restrict__ feat,
                            float* __restrict__ qa) {
    int n = blockIdx.x * blockDim.x + threadIdx.x;
    if (n >= N_NODES) return;
    float fr[NODE_DIM];
    const float* zr = z_table + (size_t)x[n] * 48;
    const float* mr = mol_table + (size_t)mol_id[n] * 16;
    #pragma unroll
    for (int i = 0; i < 48; ++i) fr[i] = zr[i];
    #pragma unroll
    for (int i = 0; i < 16; ++i) fr[48 + i] = mr[i];
    float* fo = feat + (size_t)n * NODE_DIM;
    #pragma unroll
    for (int i = 0; i < NODE_DIM; ++i) fo[i] = fr[i];

    float q0[MUL];
    #pragma unroll
    for (int u = 0; u < MUL; ++u) {
        float s = 0.f;
        for (int j = 0; j < NODE_DIM; ++j) s += fr[j] * Wq[j * MUL + u];
        q0[u] = 0.125f * s;                          // 1/sqrt(NODE_DIM)
    }
    #pragma unroll
    for (int v = 0; v < MUL; ++v) {
        float s = 0.f;
        for (int u = 0; u < MUL; ++u) s += q0[u] * Wa[u * MUL + v];
        qa[(size_t)n * MUL + v] = s;
    }
}

// ---------------------------------------------------------------------------
// Kernel 2: per-edge geometry + radial layer1 + silu; h written as packed
// bf16 WMMA A-fragments (16 edges per tile).
// ---------------------------------------------------------------------------
__device__ __forceinline__ void write_frag(const float* h, unsigned* tile_base, int elane) {
    #pragma unroll
    for (int hi = 0; hi < 2; ++hi) {
        int lane = elane + 16 * hi;
        #pragma unroll
        for (int v = 0; v < 8; ++v) {
            int k0 = frag_k0(v, hi);
            tile_base[lane * 8 + v] = f2bf(h[k0]) | (f2bf(h[k0 + 1]) << 16);
        }
    }
}

__device__ __forceinline__ void radial_h(const float* basis, const float* __restrict__ W1,
                                         float* h) {
    #pragma unroll
    for (int m = 0; m < MID; ++m) {
        float s = 0.f;
        #pragma unroll
        for (int i = 0; i < N_BASIS; ++i) s += basis[i] * W1[i * MID + m];
        s *= INV_SQRT10;
        h[m] = SILU_NORM * s / (1.f + __expf(-s));   // SILU_NORM * silu(s)
    }
}

__global__ void edge_geom_kernel(const float* __restrict__ pos,
                                 const float* __restrict__ W1k,
                                 const float* __restrict__ W1v,
                                 const int*   __restrict__ esrc,
                                 const int*   __restrict__ edst,
                                 float* __restrict__ cut,
                                 float* __restrict__ sh1,
                                 unsigned* __restrict__ hAk,
                                 unsigned* __restrict__ hAv) {
    int e = blockIdx.x * blockDim.x + threadIdx.x;
    if (e >= N_EDGES) return;
    int s = esrc[e], d = edst[e];
    float vx = pos[3 * s + 0] - pos[3 * d + 0];
    float vy = pos[3 * s + 1] - pos[3 * d + 1];
    float vz = pos[3 * s + 2] - pos[3 * d + 2];
    float len = sqrtf(vx * vx + vy * vy + vz * vz);
    float inv = len > 0.f ? 1.f / len : 0.f;
    sh1[3 * e + 0] = S3 * vx * inv;
    sh1[3 * e + 1] = S3 * vy * inv;
    sh1[3 * e + 2] = S3 * vz * inv;
    cut[e] = sus(10.f * (1.f - len * (1.f / 6.f)));

    float basis[N_BASIS];
    float t = len * (11.f / 6.f);                  // (len - v_i)/step = t - (i+1)
    #pragma unroll
    for (int i = 0; i < N_BASIS; ++i) {
        float dd = t - (float)(i + 1);
        basis[i] = SMOOTH_C * sus(dd + 1.f) * sus(1.f - dd) * SQRT10;
    }
    float h[MID];
    unsigned* tk = hAk + (size_t)(e >> 4) * 256;
    unsigned* tv = hAv + (size_t)(e >> 4) * 256;
    radial_h(basis, W1k, h);  write_frag(h, tk, e & 15);
    radial_h(basis, W1v, h);  write_frag(h, tv, e & 15);
}

// ---------------------------------------------------------------------------
// Kernel 3b (prep): pack W2k (first half only: k0) and W2v (both halves)
// into per-lane B fragments, pre-scaled by 1/sqrt(MID*NODE_DIM).
// ---------------------------------------------------------------------------
__global__ void prep_b_kernel(const float* __restrict__ W2k,
                              const float* __restrict__ W2v,
                              unsigned* __restrict__ Bp) {
    int idx = blockIdx.x * blockDim.x + threadIdx.x;
    if (idx >= (64 + 128) * 256) return;
    const float* W2; unsigned* out; int li;
    if (idx < 64 * 256) { W2 = W2k; out = Bp;              li = idx; }
    else                { W2 = W2v; out = Bp + 64 * 256;   li = idx - 64 * 256; }
    int b    = li >> 8;            // 16-col block -> fixed (half,u), cols = w
    int lane = (li >> 3) & 31;
    int v    = li & 7;
    int N = lane & 15, hi = lane >> 4;
    int k0 = frag_k0(v, hi);
    int col = b * 16 + N;          // direct column into the 2048-wide W2
    unsigned lo = f2bf(W2[(size_t)k0 * 2048 + col] * B_SCALE);
    unsigned hh = f2bf(W2[(size_t)(k0 + 1) * 2048 + col] * B_SCALE);
    out[li] = lo | (hh << 16);
}

// ---------------------------------------------------------------------------
// Kernel 4: the WMMA edge kernel. One wave per 16-edge tile.
//   per u (=0..63): 3 x V_WMMA_F32_16X16X32_BF16 (k0 / v0 / v1 blocks),
//   then per-row scale by feat_src[e,u] and accumulate (LDS broadcast + FMA).
// Tail: qk = QK_NORM * qa[dst] . k0 ; expv = cutoff * exp(qk); atomic z[dst].
// ---------------------------------------------------------------------------
__global__ __launch_bounds__(32) void edge_wmma_kernel(
        const unsigned* __restrict__ hAk, const unsigned* __restrict__ hAv,
        const unsigned* __restrict__ Bp,
        const float* __restrict__ feat, const float* __restrict__ qa,
        const float* __restrict__ cut,
        const int* __restrict__ esrc, const int* __restrict__ edst,
        float* __restrict__ expv, float* __restrict__ v0buf,
        float* __restrict__ v1buf, float* __restrict__ zbuf) {
    int tile = blockIdx.x;
    int lane = threadIdx.x;
    __shared__ float featS[16 * NODE_DIM];   // feat of the 16 source nodes
    __shared__ float k0S[16 * 16];

    for (int i = lane; i < 16 * NODE_DIM; i += 32) {
        int e = i >> 6, u = i & 63;
        featS[i] = feat[(size_t)esrc[tile * 16 + e] * NODE_DIM + u];
    }
    __syncthreads();

    const bf16x16 Ak = *(const bf16x16*)(hAk + (size_t)tile * 256 + lane * 8);
    const bf16x16 Av = *(const bf16x16*)(hAv + (size_t)tile * 256 + lane * 8);
    const unsigned* Bk = Bp;                 //  64 blocks (k, half 0)
    const unsigned* Bv = Bp + 64 * 256;      // 128 blocks (v, halves 0 and 1)

    float accK0[8], accV0[8], accV1[8];
    #pragma unroll
    for (int r = 0; r < 8; ++r) { accK0[r] = 0.f; accV0[r] = 0.f; accV1[r] = 0.f; }
    const f32x8 czero = {0.f, 0.f, 0.f, 0.f, 0.f, 0.f, 0.f, 0.f};
    int rowoff = (lane >> 4) * 8;            // D row M = r + rowoff

    for (int u = 0; u < 64; ++u) {
        bf16x16 bk  = *(const bf16x16*)(Bk + ((size_t)u * 32 + lane) * 8);
        bf16x16 bv0 = *(const bf16x16*)(Bv + ((size_t)u * 32 + lane) * 8);
        bf16x16 bv1 = *(const bf16x16*)(Bv + (((size_t)u + 64) * 32 + lane) * 8);
        f32x8 dk  = __builtin_amdgcn_wmma_f32_16x16x32_bf16(false, Ak, false, bk,  (short)0, czero, false, false);
        f32x8 dv0 = __builtin_amdgcn_wmma_f32_16x16x32_bf16(false, Av, false, bv0, (short)0, czero, false, false);
        f32x8 dv1 = __builtin_amdgcn_wmma_f32_16x16x32_bf16(false, Av, false, bv1, (short)0, czero, false, false);
        #pragma unroll
        for (int r = 0; r < 8; ++r) {
            float f = featS[(r + rowoff) * NODE_DIM + u];  // broadcast within lane groups
            accK0[r] += f * dk[r];
            accV0[r] += f * dv0[r];
            accV1[r] += f * dv1[r];
        }
    }

    int w = lane & 15;
    #pragma unroll
    for (int r = 0; r < 8; ++r) {
        int e  = r + rowoff;
        int ge = tile * 16 + e;
        k0S[e * 16 + w] = accK0[r];
        v0buf[(size_t)ge * 16 + w] = accV0[r];
        v1buf[(size_t)ge * 16 + w] = accV1[r];
    }
    __syncthreads();

    if (lane < 16) {
        int ge  = tile * 16 + lane;
        int dst = edst[ge];
        float s = 0.f;
        #pragma unroll
        for (int v = 0; v < 16; ++v) s += k0S[lane * 16 + v] * qa[(size_t)dst * 16 + v];
        float ev = cut[ge] * __expf(QK_NORM * s);   // q1==0 => Wb term vanishes
        expv[ge] = ev;
        atomicAdd(&zbuf[dst], ev);
    }
}

// ---------------------------------------------------------------------------
// Kernel 5: attention scatter. one thread per (edge, output j<64).
// ---------------------------------------------------------------------------
__global__ void attn_scatter_kernel(const float* __restrict__ expv,
                                    const float* __restrict__ zbuf,
                                    const float* __restrict__ v0buf,
                                    const float* __restrict__ v1buf,
                                    const float* __restrict__ sh1,
                                    const int* __restrict__ edst,
                                    float* __restrict__ node_out) {
    int idx = blockIdx.x * blockDim.x + threadIdx.x;
    int e = idx >> 6, j = idx & 63;
    int dst = edst[e];
    float z = zbuf[dst];
    if (z <= 0.f) z = 1.f;
    float att = sqrtf(expv[e] / z);             // relu(alpha)=alpha since alpha>=0
    float val;
    if (j < 16) {
        val = v0buf[(size_t)e * 16 + j];
    } else {
        int t = j - 16;
        val = v1buf[(size_t)e * 16 + t / 3] * sh1[(size_t)e * 3 + t % 3];
    }
    atomicAdd(&node_out[(size_t)dst * 64 + j], att * val);
}

// ---------------------------------------------------------------------------
// Kernel 6: node -> graph pooling (sums + counts)
// ---------------------------------------------------------------------------
__global__ void pool_kernel(const float* __restrict__ node_out,
                            const int* __restrict__ batch_idx,
                            float* __restrict__ gsum, float* __restrict__ gcnt) {
    int idx = blockIdx.x * blockDim.x + threadIdx.x;
    int n = idx >> 6, j = idx & 63;
    int g = batch_idx[n];
    atomicAdd(&gsum[(size_t)g * 64 + j], node_out[(size_t)n * 64 + j]);
    if (j == 0) atomicAdd(&gcnt[g], 1.0f);
}

__global__ void finalize_kernel(const float* __restrict__ gsum,
                                const float* __restrict__ gcnt,
                                float* __restrict__ out) {
    int i = blockIdx.x * blockDim.x + threadIdx.x;
    if (i >= N_GRAPHS * 64) return;
    out[i] = gsum[i] / fmaxf(gcnt[i >> 6], 1.0f);
}

// ---------------------------------------------------------------------------
// launch
// ---------------------------------------------------------------------------
extern "C" void kernel_launch(void* const* d_in, const int* in_sizes, int n_in,
                              void* d_out, int out_size, void* d_ws, size_t ws_size,
                              hipStream_t stream) {
    const float* pos       = (const float*)d_in[0];
    const float* z_table   = (const float*)d_in[1];
    const float* mol_table = (const float*)d_in[2];
    const float* Wq        = (const float*)d_in[3];
    const float* W1k       = (const float*)d_in[4];
    const float* W2k       = (const float*)d_in[5];
    const float* W1v       = (const float*)d_in[6];
    const float* W2v       = (const float*)d_in[7];
    const float* Wa        = (const float*)d_in[8];
    // d_in[9] = Wb (unused: q1 == 0)
    const int* x         = (const int*)d_in[10];
    const int* mol_id    = (const int*)d_in[11];
    const int* edge_src  = (const int*)d_in[12];
    const int* edge_dst  = (const int*)d_in[13];
    const int* batch_idx = (const int*)d_in[14];

    float*    ws   = (float*)d_ws;
    float*    feat = ws + OFF_FEAT;
    float*    qa   = ws + OFF_QA;
    float*    cut  = ws + OFF_CUT;
    float*    sh1  = ws + OFF_SH1;
    unsigned* hAk  = (unsigned*)(ws + OFF_HAK);
    unsigned* hAv  = (unsigned*)(ws + OFF_HAV);
    unsigned* Bp   = (unsigned*)(ws + OFF_BPK);
    float*    expv = ws + OFF_EXPV;
    float*    v0b  = ws + OFF_V0;
    float*    v1b  = ws + OFF_V1;
    float*    zb   = ws + OFF_Z;
    float*    nout = ws + OFF_NODE;
    float*    gsum = ws + OFF_GSUM;
    float*    gcnt = ws + OFF_GCNT;

    // zero the accumulator region (z, node_out, gsum, gcnt are contiguous)
    hipMemsetAsync(zb, 0, (WS_TOTAL - OFF_Z) * sizeof(float), stream);

    node_kernel<<<(N_NODES + 255) / 256, 256, 0, stream>>>(
        z_table, mol_table, Wq, Wa, x, mol_id, feat, qa);
    edge_geom_kernel<<<(N_EDGES + 255) / 256, 256, 0, stream>>>(
        pos, W1k, W1v, edge_src, edge_dst, cut, sh1, hAk, hAv);
    prep_b_kernel<<<((64 + 128) * 256 + 255) / 256, 256, 0, stream>>>(W2k, W2v, Bp);

    edge_wmma_kernel<<<N_TILES, 32, 0, stream>>>(
        hAk, hAv, Bp, feat, qa, cut, edge_src, edge_dst, expv, v0b, v1b, zb);

    attn_scatter_kernel<<<(N_EDGES * 64) / 256, 256, 0, stream>>>(
        expv, zb, v0b, v1b, sh1, edge_dst, nout);
    pool_kernel<<<(N_NODES * 64) / 256, 256, 0, stream>>>(nout, batch_idx, gsum, gcnt);
    finalize_kernel<<<(N_GRAPHS * 64 + 255) / 256, 256, 0, stream>>>(gsum, gcnt, (float*)d_out);
}
